// SAMMCodebook_80616536145995
// MI455X (gfx1250) — matrix operand order repaired and checked by reading, not compiled, not run on hardware
//
#include <hip/hip_runtime.h>
#include <stdint.h>

typedef __bf16 bf16_t;
typedef bf16_t v16bf __attribute__((ext_vector_type(16)));
typedef float  v8f   __attribute__((ext_vector_type(8)));
typedef float  v4f   __attribute__((ext_vector_type(4)));   // native vec for NT loads

__device__ __forceinline__ uint32_t f32_to_bf16_rne(float f) {
    union { float f; uint32_t u; } cv;
    cv.f = f;
    uint32_t u = cv.u;
    return (u + 0x7FFFu + ((u >> 16) & 1u)) >> 16;   // round-to-nearest-even
}

// ---------------------------------------------------------------------------
// Prep: codebook f32 -> bf16 (rows contiguous == WMMA B columns) + c_sq[k].
// ---------------------------------------------------------------------------
__global__ void samm_prep_codebook(const float* __restrict__ cb,
                                   uint16_t* __restrict__ cb_bf,
                                   float* __restrict__ c_sq, int D) {
    const int k = blockIdx.x;
    const float* row = cb + (size_t)k * D;
    uint16_t* orow = cb_bf + (size_t)k * D;
    float s = 0.0f;
    for (int d = threadIdx.x; d < D; d += blockDim.x) {
        float v = row[d];
        s += v * v;
        orow[d] = (uint16_t)f32_to_bf16_rne(v);
    }
    __shared__ float red[256];
    red[threadIdx.x] = s;
    __syncthreads();
    for (int off = 128; off > 0; off >>= 1) {
        if ((int)threadIdx.x < off) red[threadIdx.x] += red[threadIdx.x + off];
        __syncthreads();
    }
    if (threadIdx.x == 0) c_sq[k] = red[0];
}

// ---------------------------------------------------------------------------
// Main kernel: 256 threads (8 waves, wave32). Workgroup = 64 rows x 512 cols.
// Wave w owns columns [w*64, w*64+64): 4 N-tiles x 4 M-tiles of 16x16.
// D walked 64 at a time (2 WMMA sub-chunks), double-buffered LDS h tile.
// ---------------------------------------------------------------------------
#define LDS_ROW_U32 20   // 80-byte row stride (32 bf16 data + 16B pad)

union BFrag { v16bf v; uint4 q[2]; };

// Stage one 64x32 f32 sub-tile of h into LDS as packed bf16 pairs.
// Non-temporal loads: h streams through exactly once; keep L2 for codebook.
__device__ __forceinline__ void stage_tile(uint32_t* __restrict__ dst_base,
                                           const float* __restrict__ h,
                                           long long row0, int D, int d0,
                                           int tid) {
    const int r   = tid >> 2;      // 0..63
    const int seg = tid & 3;       // 8 floats each
    const v4f* src =
        (const v4f*)(h + (row0 + r) * (long long)D + d0 + seg * 8);
    v4f a = __builtin_nontemporal_load(src);
    v4f b = __builtin_nontemporal_load(src + 1);
    uint32_t* dst = dst_base + r * LDS_ROW_U32 + seg * 4;
    dst[0] = f32_to_bf16_rne(a.x) | (f32_to_bf16_rne(a.y) << 16);
    dst[1] = f32_to_bf16_rne(a.z) | (f32_to_bf16_rne(a.w) << 16);
    dst[2] = f32_to_bf16_rne(b.x) | (f32_to_bf16_rne(b.y) << 16);
    dst[3] = f32_to_bf16_rne(b.z) | (f32_to_bf16_rne(b.w) << 16);
}

__global__ __launch_bounds__(256)
void samm_encode_kernel(const float* __restrict__ h,
                        const uint16_t* __restrict__ cb_bf,
                        const float* __restrict__ c_sq,
                        int* __restrict__ out, int D) {
    // [buffer][sub-chunk][64 rows * 20 dwords]  -> 20 KB total
    __shared__ __align__(16) uint32_t shA[2][2][64 * LDS_ROW_U32];
    __shared__ unsigned long long smin[64];

    const int tid  = threadIdx.x;
    const int wave = tid >> 5;
    const int lane = tid & 31;
    const int l15  = lane & 15;
    const int hi   = lane >> 4;           // 0: lanes 0-15, 1: lanes 16-31
    const long long row0 = (long long)blockIdx.x * 64;

    if (tid < 64) smin[tid] = ~0ull;

    v8f acc[4][4];
    #pragma unroll
    for (int m = 0; m < 4; ++m)
        #pragma unroll
        for (int n = 0; n < 4; ++n)
            #pragma unroll
            for (int i = 0; i < 8; ++i) acc[m][n][i] = 0.0f;

    const int colBase = wave * 64;

    // Preload first 64-wide D chunk into buffer 0.
    stage_tile(&shA[0][0][0], h, row0, D, 0, tid);
    stage_tile(&shA[0][1][0], h, row0, D, 32, tid);

    int buf = 0;
    for (int d0 = 0; d0 < D; d0 += 64) {
        __syncthreads();   // staging of `buf` visible; prior reads of buf^1 done

        // Issue next chunk's staging early; drains while WMMAs run.
        if (d0 + 64 < D) {
            stage_tile(&shA[buf ^ 1][0][0], h, row0, D, d0 + 64, tid);
            stage_tile(&shA[buf ^ 1][1][0], h, row0, D, d0 + 96, tid);
        }

        #pragma unroll
        for (int cc = 0; cc < 2; ++cc) {
            const int dc = d0 + cc * 32;

            // B fragments: 4 column tiles straight from bf16 codebook (L2-hot).
            BFrag bfrag[4];
            #pragma unroll
            for (int n = 0; n < 4; ++n) {
                const int c = colBase + n * 16 + l15;
                const char* p =
                    (const char*)cb_bf + ((size_t)c * D + dc) * 2 + hi * 32;
                bfrag[n].q[0] = *(const uint4*)(p);
                bfrag[n].q[1] = *(const uint4*)(p + 16);
            }

            // A fragments from LDS + 16 WMMAs.
            #pragma unroll
            for (int m = 0; m < 4; ++m) {
                BFrag afrag;
                const uint32_t* arow =
                    &shA[buf][cc][(m * 16 + l15) * LDS_ROW_U32];
                // lo lanes: K {0..15}; hi lanes: K {16..31}
                afrag.q[0] = *(const uint4*)(arow + hi * 8);
                afrag.q[1] = *(const uint4*)(arow + hi * 8 + 4);
                #pragma unroll
                for (int n = 0; n < 4; ++n) {
                    acc[m][n] = __builtin_amdgcn_wmma_f32_16x16x32_bf16(
                        /*neg_a=*/false, afrag.v,
                        /*neg_b=*/false, bfrag[n].v,
                        /*c_mod=*/(short)0, acc[m][n],
                        /*reuse_a=*/false, /*reuse_b=*/false);
                }
            }
        }
        buf ^= 1;
    }

    // ---- epilogue: dist = c_sq - 2*dot, argmin per row ----
    float cs[4];
    #pragma unroll
    for (int n = 0; n < 4; ++n) cs[n] = c_sq[colBase + n * 16 + l15];

    #pragma unroll
    for (int m = 0; m < 4; ++m) {
        #pragma unroll
        for (int v = 0; v < 8; ++v) {
            const int rloc = m * 16 + v + hi * 8;
            float best = cs[0] - 2.0f * acc[m][0][v];
            int   bcol = colBase + l15;
            #pragma unroll
            for (int n = 1; n < 4; ++n) {
                float dv = cs[n] - 2.0f * acc[m][n][v];
                int   c  = colBase + n * 16 + l15;
                if (dv < best) { best = dv; bcol = c; }  // cols ascend: ties keep lowest
            }
            uint32_t fb  = __float_as_uint(best);
            uint32_t key = fb ^ ((uint32_t)((int32_t)fb >> 31) | 0x80000000u);
            unsigned long long packed =
                ((unsigned long long)key << 32) | (uint32_t)bcol;
            // reduce across the 16 lanes sharing this row (stays in half-wave)
            #pragma unroll
            for (int off = 1; off < 16; off <<= 1) {
                unsigned long long other = __shfl_xor(packed, off, 32);
                if (other < packed) packed = other;
            }
            if (l15 == 0) atomicMin(&smin[rloc], packed);   // ds_min_u64
        }
    }
    __syncthreads();
    if (tid < 64) {
        int idx = (int)(smin[tid] & 0xFFFFFFFFu);
        __builtin_nontemporal_store(idx, out + row0 + tid);
    }
}

// ---------------------------------------------------------------------------
extern "C" void kernel_launch(void* const* d_in, const int* in_sizes, int n_in,
                              void* d_out, int out_size, void* d_ws, size_t ws_size,
                              hipStream_t stream) {
    const float* h  = (const float*)d_in[0];   // [B,T,D] f32
    const float* cb = (const float*)d_in[1];   // [K,D]   f32
    const int N = out_size;                    // 65536
    const int D = in_sizes[0] / N;             // 1024
    const int K = in_sizes[1] / D;             // 512

    uint16_t* cb_bf = (uint16_t*)d_ws;                               // K*D*2 bytes
    float*    c_sq  = (float*)((char*)d_ws + (size_t)K * D * 2);     // K*4 bytes

    samm_prep_codebook<<<K, 256, 0, stream>>>(cb, cb_bf, c_sq, D);
    samm_encode_kernel<<<N / 64, 256, 0, stream>>>(h, cb_bf, c_sq, (int*)d_out, D);
}
